// MINCUTPOOL_AE_32392643346837
// MI455X (gfx1250) — matrix operand order repaired and checked by reading, not compiled, not run on hardware
//
#include <hip/hip_runtime.h>
#include <hip/hip_bf16.h>

typedef float v2f  __attribute__((ext_vector_type(2)));
typedef float v8f  __attribute__((ext_vector_type(8)));
typedef __bf16 bhalf;
typedef bhalf v8bf  __attribute__((ext_vector_type(8)));
typedef bhalf v16bf __attribute__((ext_vector_type(16)));

#define EPS_ 1e-15f

// ---------------- graph prep ----------------
__global__ void k_deg(const int* row, const int* col, float* degc, float* degrow, int E) {
    int e = blockIdx.x * blockDim.x + threadIdx.x;
    if (e >= E) return;
    atomicAdd(&degc[col[e]], 1.0f);
    atomicAdd(&degrow[row[e]], 1.0f);
}

__global__ void k_dinv(const float* degc, float* dinv, int N) {
    int i = blockIdx.x * blockDim.x + threadIdx.x;
    if (i >= N) return;
    dinv[i] = rsqrtf(degc[i] + 1.0f);  // +1 self loop, always > 0
}

__global__ void k_norm(const int* row, const int* col, const float* dinv, float* norm, int E) {
    int e = blockIdx.x * blockDim.x + threadIdx.x;
    if (e >= E) return;
    norm[e] = dinv[row[e]] * dinv[col[e]];
}

// dst[didx[e], :] += scale[e] * src[sidx[e], :]   (one wave per edge)
__global__ void k_scatter(const float* __restrict__ src, float* __restrict__ dst,
                          const int* sidx, const int* didx, const float* scale,
                          int E, int W) {
    int wave = (blockIdx.x * blockDim.x + threadIdx.x) >> 5;
    int lane = threadIdx.x & 31;
    if (wave >= E) return;
    int sr = sidx[wave], dr = didx[wave];
    float sc = scale ? scale[wave] : 1.0f;
    const float* sp = src + (size_t)sr * W;
    float* dp = dst + (size_t)dr * W;
    for (int j = lane; j < W; j += 32) atomicAdd(&dp[j], sc * sp[j]);
}

// out = act(agg + h * dinv^2 (self loop) + bias)
__global__ void k_gcn_finish(const float* agg, const float* h, const float* dinv,
                             const float* bias, float* out, int N, int W, int act) {
    size_t t = (size_t)blockIdx.x * blockDim.x + threadIdx.x;
    if (t >= (size_t)N * W) return;
    int i = (int)(t / W), j = (int)(t % W);
    float dv = dinv[i];
    float v = agg[t] + h[t] * dv * dv + bias[j];
    out[t] = act ? tanhf(v) : v;
}

// ---------------- fp32 WMMA GEMM (NN): C[M,Nc] = A[M,Kd] @ B[Kd,Nc] ----------------
// one wave per 16x64 strip of C (4 x 16x16 tiles, A fragment reused). Kd % 4 == 0.
// All loads are index-clamped instead of masked: an OOB A-row / B-col only pollutes
// C rows/cols that are never stored. Inner loop is branch-free (EXEC untouched).
__global__ void k_gemm_f32(const float* __restrict__ A, const float* __restrict__ B,
                           float* __restrict__ C, int M, int Kd, int Nc,
                           const float* bias, int act) {
    int lane = threadIdx.x & 31;
    int l15 = lane & 15;
    int half = lane >> 4;        // 0/1
    int kh = half << 1;          // 0/2
    int wave = blockIdx.x * (blockDim.x >> 5) + (threadIdx.x >> 5);
    int tilesM = (M + 15) >> 4;
    int strips = (Nc + 63) >> 6;
    if (wave >= tilesM * strips) return;
    int tm = wave / strips, st = wave % strips;
    int m0 = tm << 4, n0 = st << 6;

    int mrow = m0 + l15; if (mrow >= M) mrow = M - 1;   // clamp: safe unguarded load
    const float* Ap = A + (size_t)mrow * Kd + kh;

    int nc0 = n0 + l15;       if (nc0 >= Nc) nc0 = Nc - 1;
    int nc1 = n0 + 16 + l15;  if (nc1 >= Nc) nc1 = Nc - 1;
    int nc2 = n0 + 32 + l15;  if (nc2 >= Nc) nc2 = Nc - 1;
    int nc3 = n0 + 48 + l15;  if (nc3 >= Nc) nc3 = Nc - 1;
    const float* B0 = B + (size_t)kh * Nc + nc0;
    const float* B1 = B + (size_t)kh * Nc + nc1;
    const float* B2 = B + (size_t)kh * Nc + nc2;
    const float* B3 = B + (size_t)kh * Nc + nc3;
    size_t bstep = (size_t)4 * Nc;

    v8f acc0 = (v8f){0.f, 0.f, 0.f, 0.f, 0.f, 0.f, 0.f, 0.f};
    v8f acc1 = acc0, acc2 = acc0, acc3 = acc0;

#pragma unroll 2
    for (int k = 0; k < Kd; k += 4) {
        v2f a = *(const v2f*)Ap;
        Ap += 4;
        v2f b0; b0.x = B0[0]; b0.y = B0[Nc];
        v2f b1; b1.x = B1[0]; b1.y = B1[Nc];
        v2f b2; b2.x = B2[0]; b2.y = B2[Nc];
        v2f b3; b3.x = B3[0]; b3.y = B3[Nc];
        B0 += bstep; B1 += bstep; B2 += bstep; B3 += bstep;
        acc0 = __builtin_amdgcn_wmma_f32_16x16x4_f32(false, a, false, b0, (short)0, acc0, false, false);
        acc1 = __builtin_amdgcn_wmma_f32_16x16x4_f32(false, a, false, b1, (short)0, acc1, false, false);
        acc2 = __builtin_amdgcn_wmma_f32_16x16x4_f32(false, a, false, b2, (short)0, acc2, false, false);
        acc3 = __builtin_amdgcn_wmma_f32_16x16x4_f32(false, a, false, b3, (short)0, acc3, false, false);
    }

    int mb = m0 + (half << 3);
    v8f accs[4] = {acc0, acc1, acc2, acc3};
#pragma unroll
    for (int t = 0; t < 4; ++t) {
        int ncs = n0 + 16 * t + l15;
        if (ncs < Nc) {
            float bz = bias ? bias[ncs] : 0.f;
#pragma unroll
            for (int r = 0; r < 8; ++r) {
                int m = mb + r;
                if (m < M) {
                    float v = accs[t][r] + bz;
                    if (act) v = tanhf(v);
                    C[(size_t)m * Nc + ncs] = v;
                }
            }
        }
    }
}

// ---------------- bf16 WMMA GEMM (NT): C[M,Nc] = A[M,Kp] @ B[Nc,Kp]^T ----------------
// M % 16 == 0, Nc % 16 == 0, Kp % 32 == 0. one wave per 16x64 strip of C.
// B rows clamped -> all 4 subtiles computed unconditionally; stores guarded.
__global__ void k_gemm_bf16_nt(const bhalf* __restrict__ A, const bhalf* __restrict__ B,
                               float* __restrict__ C, int M, int Nc, int Kp) {
    int lane = threadIdx.x & 31;
    int l15 = lane & 15;
    int half = lane >> 4;
    int wave = blockIdx.x * (blockDim.x >> 5) + (threadIdx.x >> 5);
    int tilesM = M >> 4;
    int strips = (Nc + 63) >> 6;
    if (wave >= tilesM * strips) return;
    int tm = wave / strips, st = wave % strips;
    int m0 = tm << 4, n0 = st << 6;

    int ach = half << 3;   // A chunk: 0 or 8
    int bk = half << 3;    // B k-offset: 0 or 8
    const bhalf* Ap = A + (size_t)(m0 + l15) * Kp + ach;

    int nc0 = n0 + l15;       if (nc0 >= Nc) nc0 = Nc - 1;
    int nc1 = n0 + 16 + l15;  if (nc1 >= Nc) nc1 = Nc - 1;
    int nc2 = n0 + 32 + l15;  if (nc2 >= Nc) nc2 = Nc - 1;
    int nc3 = n0 + 48 + l15;  if (nc3 >= Nc) nc3 = Nc - 1;
    const bhalf* B0 = B + (size_t)nc0 * Kp + bk;
    const bhalf* B1 = B + (size_t)nc1 * Kp + bk;
    const bhalf* B2 = B + (size_t)nc2 * Kp + bk;
    const bhalf* B3 = B + (size_t)nc3 * Kp + bk;

    v8f acc0 = (v8f){0.f, 0.f, 0.f, 0.f, 0.f, 0.f, 0.f, 0.f};
    v8f acc1 = acc0, acc2 = acc0, acc3 = acc0;

#pragma unroll 2
    for (int k = 0; k < Kp; k += 32) {
        v8bf a0 = *(const v8bf*)(Ap + k);
        v8bf a1 = *(const v8bf*)(Ap + k + 16);
        v16bf a = __builtin_shufflevector(a0, a1, 0, 1, 2, 3, 4, 5, 6, 7,
                                          8, 9, 10, 11, 12, 13, 14, 15);
        v16bf b0 = __builtin_shufflevector(*(const v8bf*)(B0 + k), *(const v8bf*)(B0 + k + 16), 0, 1, 2, 3, 4, 5, 6, 7, 8, 9, 10, 11, 12, 13, 14, 15);
        v16bf b1 = __builtin_shufflevector(*(const v8bf*)(B1 + k), *(const v8bf*)(B1 + k + 16), 0, 1, 2, 3, 4, 5, 6, 7, 8, 9, 10, 11, 12, 13, 14, 15);
        v16bf b2 = __builtin_shufflevector(*(const v8bf*)(B2 + k), *(const v8bf*)(B2 + k + 16), 0, 1, 2, 3, 4, 5, 6, 7, 8, 9, 10, 11, 12, 13, 14, 15);
        v16bf b3 = __builtin_shufflevector(*(const v8bf*)(B3 + k), *(const v8bf*)(B3 + k + 16), 0, 1, 2, 3, 4, 5, 6, 7, 8, 9, 10, 11, 12, 13, 14, 15);
        acc0 = __builtin_amdgcn_wmma_f32_16x16x32_bf16(false, a, false, b0, (short)0, acc0, false, false);
        acc1 = __builtin_amdgcn_wmma_f32_16x16x32_bf16(false, a, false, b1, (short)0, acc1, false, false);
        acc2 = __builtin_amdgcn_wmma_f32_16x16x32_bf16(false, a, false, b2, (short)0, acc2, false, false);
        acc3 = __builtin_amdgcn_wmma_f32_16x16x32_bf16(false, a, false, b3, (short)0, acc3, false, false);
    }

    int mb = m0 + (half << 3);
    v8f accs[4] = {acc0, acc1, acc2, acc3};
#pragma unroll
    for (int t = 0; t < 4; ++t) {
        int ncs = n0 + 16 * t + l15;
        if (ncs < Nc) {
#pragma unroll
            for (int r = 0; r < 8; ++r)
                C[(size_t)(mb + r) * Nc + ncs] = accs[t][r];
        }
    }
}

// ---------------- softmax (one block per row, blockDim=512, K<=512) ----------------
// in-place softmax on slog row; also writes bf16 padded copy; accumulates mincut denom
__global__ void k_softmax(float* slog, bhalf* s_bf, const float* degrow,
                          float* acc, int K, int Kp) {
    __shared__ float red[512];
    int rowi = blockIdx.x;
    int t = threadIdx.x;
    float* rp = slog + (size_t)rowi * K;
    float v = (t < K) ? rp[t] : -3.4e38f;
    red[t] = v; __syncthreads();
    for (int s = 256; s > 0; s >>= 1) { if (t < s) red[t] = fmaxf(red[t], red[t + s]); __syncthreads(); }
    float mx = red[0]; __syncthreads();
    float e = (t < K) ? __expf(v - mx) : 0.f;
    red[t] = e; __syncthreads();
    for (int s = 256; s > 0; s >>= 1) { if (t < s) red[t] += red[t + s]; __syncthreads(); }
    float tot = red[0]; __syncthreads();
    float p = 0.f;
    if (t < K) { p = e / tot; rp[t] = p; }
    s_bf[(size_t)rowi * Kp + t] = (bhalf)p;
    red[t] = p * p; __syncthreads();
    for (int s = 256; s > 0; s >>= 1) { if (t < s) red[t] += red[t + s]; __syncthreads(); }
    if (t == 0) atomicAdd(&acc[0], degrow[rowi] * red[0]);
}

// ---------------- transpose: out[c,r] = in[r,c] ----------------
__global__ void k_transpose(const float* in, float* out, int rows, int cols) {
    __shared__ float tile[32][33];
    int c = blockIdx.x * 32 + threadIdx.x;
    int rb = blockIdx.y * 32;
    for (int dy = threadIdx.y; dy < 32; dy += 8) {
        int r = rb + dy;
        tile[dy][threadIdx.x] = (r < rows && c < cols) ? in[(size_t)r * cols + c] : 0.f;
    }
    __syncthreads();
    int rr = rb + threadIdx.x;
    int cb = blockIdx.x * 32;
    for (int dy = threadIdx.y; dy < 32; dy += 8) {
        int cc = cb + dy;
        if (cc < cols && rr < rows) out[(size_t)cc * rows + rr] = tile[threadIdx.x][dy];
    }
}

// ---------------- loss reductions ----------------
__global__ void k_reduce_ss(const float* ss, float* acc, int K) {
    __shared__ float sq[256], tr[256];
    int t = blockIdx.x * 256 + threadIdx.x;
    float vq = 0.f, vt = 0.f;
    if (t < K * K) {
        float v = ss[t];
        vq = v * v;
        if (t / K == t % K) vt = v;
    }
    sq[threadIdx.x] = vq; tr[threadIdx.x] = vt; __syncthreads();
    for (int s = 128; s > 0; s >>= 1) {
        if (threadIdx.x < s) { sq[threadIdx.x] += sq[threadIdx.x + s]; tr[threadIdx.x] += tr[threadIdx.x + s]; }
        __syncthreads();
    }
    if (threadIdx.x == 0) { atomicAdd(&acc[2], sq[0]); atomicAdd(&acc[3], tr[0]); }
}

// per-row off-diagonal sum of out_adj -> d = sqrt(rowsum)+EPS ; accumulate trace
__global__ void k_oa_rowsum(const float* oa, float* dvec, float* acc, int K) {
    __shared__ float red[512];
    int i = blockIdx.x;
    int t = threadIdx.x;
    float v = (t < K && t != i) ? oa[(size_t)i * K + t] : 0.f;
    red[t] = v; __syncthreads();
    for (int s = 256; s > 0; s >>= 1) { if (t < s) red[t] += red[t + s]; __syncthreads(); }
    if (t == 0) {
        dvec[i] = sqrtf(red[0]) + EPS_;
        atomicAdd(&acc[1], oa[(size_t)i * K + i]);
    }
}

__global__ void k_oa_norm(float* oa, const float* dvec, int K) {
    int t = blockIdx.x * blockDim.x + threadIdx.x;
    if (t >= K * K) return;
    int i = t / K, j = t % K;
    oa[t] = (i == j) ? 0.f : oa[t] / (dvec[i] * dvec[j]);
}

__global__ void k_tobf16pad(const float* in, bhalf* out, int N, int K, int Kp) {
    size_t t = (size_t)blockIdx.x * blockDim.x + threadIdx.x;
    if (t >= (size_t)N * Kp) return;
    int j = (int)(t % Kp);
    int i = (int)(t / Kp);
    out[t] = (j < K) ? (bhalf)in[(size_t)i * K + j] : (bhalf)0.f;
}

__global__ void k_finalize(const float* acc, float* out, int K) {
    float den = acc[0] + EPS_;
    out[0] = -(acc[1] / den);
    float ssn = sqrtf(acc[2]);
    float a = 1.f / (ssn + EPS_);
    float o2 = acc[2] * a * a - 2.f * a * acc[3] / sqrtf((float)K) + 1.f;
    out[1] = sqrtf(fmaxf(o2, 0.f));
}

// ---------------- host ----------------
extern "C" void kernel_launch(void* const* d_in, const int* in_sizes, int n_in,
                              void* d_out, int out_size, void* d_ws, size_t ws_size,
                              hipStream_t stream) {
    const int F = 128, H = 128, K = 500, KP = 512;
    const int N = in_sizes[0] / F;
    const int E = in_sizes[1] / 2;

    const float* nodes = (const float*)d_in[0];
    const int* row = (const int*)d_in[1];
    const int* col = row + E;
    const float* W1 = (const float*)d_in[3];  const float* b1 = (const float*)d_in[4];
    const float* W2 = (const float*)d_in[5];  const float* b2 = (const float*)d_in[6];
    const float* poolW = (const float*)d_in[7]; const float* poolb = (const float*)d_in[8];
    const float* W3 = (const float*)d_in[9];  const float* b3 = (const float*)d_in[10];
    const float* W4 = (const float*)d_in[11]; const float* b4 = (const float*)d_in[12];
    const float* W5 = (const float*)d_in[13]; const float* b5 = (const float*)d_in[14];

    float* out_x = (float*)d_out;
    float* out_adj_big = out_x + (size_t)N * F;
    float* out_loss = out_adj_big + (size_t)N * N;

    float* p = (float*)d_ws;
    auto alloc = [&](size_t n) { float* r = p; p += n; return r; };
    float* degc   = alloc(N);
    float* degrow = alloc(N);
    float* dinv   = alloc(N);
    float* norm   = alloc(E);
    float* h      = alloc((size_t)N * H);
    float* agg    = alloc((size_t)N * H);
    float* xA     = alloc((size_t)N * H);
    float* xB     = alloc((size_t)N * H);
    float* xOut   = alloc((size_t)N * H);
    float* s      = alloc((size_t)N * K);
    float* sT     = alloc((size_t)N * K);
    float* a_s    = alloc((size_t)N * K);       // reused later as u
    bhalf* s_bf   = (bhalf*)alloc((size_t)N * KP / 2);
    bhalf* u_bf   = (bhalf*)alloc((size_t)N * KP / 2);
    float* outKH  = alloc((size_t)K * H);
    float* oadj   = alloc((size_t)K * K);
    float* oadjT  = alloc((size_t)K * K);
    float* ss     = alloc((size_t)K * K);
    float* dvec   = alloc(K);
    float* acc    = alloc(8);
    (void)ws_size; (void)n_in; (void)out_size;

    auto gemm = [&](const float* A, const float* B, float* C, int M, int Kd, int Nc,
                    const float* bias, int act) {
        int tilesM = (M + 15) / 16;
        int strips = (Nc + 63) / 64;
        int waves = tilesM * strips;
        int blocks = (waves + 7) / 8;
        k_gemm_f32<<<blocks, 256, 0, stream>>>(A, B, C, M, Kd, Nc, bias, act);
    };
    auto gcn = [&](const float* xin, const float* W, const float* b, float* xout, int act) {
        gemm(xin, W, h, N, H, H, nullptr, 0);
        hipMemsetAsync(agg, 0, sizeof(float) * (size_t)N * H, stream);
        k_scatter<<<(E + 7) / 8, 256, 0, stream>>>(h, agg, row, col, norm, E, H);
        int nb = (int)(((size_t)N * H + 255) / 256);
        k_gcn_finish<<<nb, 256, 0, stream>>>(agg, h, dinv, b, xout, N, H, act);
    };

    hipMemsetAsync(degc, 0, sizeof(float) * 2 * (size_t)N, stream);  // degc + degrow
    hipMemsetAsync(acc, 0, sizeof(float) * 8, stream);

    k_deg<<<(E + 255) / 256, 256, 0, stream>>>(row, col, degc, degrow, E);
    k_dinv<<<(N + 255) / 256, 256, 0, stream>>>(degc, dinv, N);
    k_norm<<<(E + 255) / 256, 256, 0, stream>>>(row, col, dinv, norm, E);

    gcn(nodes, W1, b1, xA, 1);
    gcn(xA, W2, b2, xB, 1);

    gemm(xB, poolW, s, N, H, K, poolb, 0);                 // s_logits
    k_softmax<<<N, 512, 0, stream>>>(s, s_bf, degrow, acc, K, KP);
    k_transpose<<<dim3((K + 31) / 32, (N + 31) / 32), dim3(32, 8), 0, stream>>>(s, sT, N, K);

    hipMemsetAsync(a_s, 0, sizeof(float) * (size_t)N * K, stream);
    k_scatter<<<(E + 7) / 8, 256, 0, stream>>>(s, a_s, col, row, nullptr, E, K);  // a_s[row]+=s[col]

    gemm(sT, xB, outKH, K, N, H, nullptr, 0);              // out = s^T x
    gemm(sT, a_s, oadj, K, N, K, nullptr, 0);              // out_adj = s^T A s
    gemm(sT, s, ss, K, N, K, nullptr, 0);                  // ss = s^T s
    k_reduce_ss<<<(K * K + 255) / 256, 256, 0, stream>>>(ss, acc, K);
    k_oa_rowsum<<<K, 512, 0, stream>>>(oadj, dvec, acc, K);
    k_oa_norm<<<(K * K + 255) / 256, 256, 0, stream>>>(oadj, dvec, K);
    k_transpose<<<dim3((K + 31) / 32, (K + 31) / 32), dim3(32, 8), 0, stream>>>(oadj, oadjT, K, K);

    gemm(s, outKH, xOut, N, K, H, nullptr, 0);             // x_out = s @ out
    gemm(s, oadjT, a_s, N, K, K, nullptr, 0);              // u = s @ out_adj^T  (reuses a_s)
    {
        int nb = (int)(((size_t)N * KP + 255) / 256);
        k_tobf16pad<<<nb, 256, 0, stream>>>(a_s, u_bf, N, K, KP);
    }
    {
        int tilesM = N / 16;
        int strips = (N + 63) / 64;
        int waves = tilesM * strips;
        int blocks = (waves + 7) / 8;
        k_gemm_bf16_nt<<<blocks, 256, 0, stream>>>(s_bf, u_bf, out_adj_big, N, N, KP);
    }

    gcn(xOut, W3, b3, xA, 1);
    gcn(xA, W4, b4, xB, 1);
    gcn(xB, W5, b5, out_x, 0);

    k_finalize<<<1, 1, 0, stream>>>(acc, out_loss, K);
}
